// SimpleMaxViTBlock_5282809774630
// MI455X (gfx1250) — matrix operand order repaired and checked by reading, not compiled, not run on hardware
//
#include <hip/hip_runtime.h>

typedef __attribute__((ext_vector_type(16))) _Float16 v16h;
typedef __attribute__((ext_vector_type(8)))  float    v8f;

#define LDSTR 40   // LDS row stride in halfs (32 + 8 pad) -> conflict-light

// ---- CDNA5 async global->LDS DMA (ISA 15.18.3 op 98, ASYNCcnt-tracked) ----
__device__ __forceinline__ void async_load_b128(unsigned lds_off, const void* gaddr) {
  asm volatile("global_load_async_to_lds_b128 %0, %1, off"
               :: "v"(lds_off), "v"((unsigned long long)gaddr)
               : "memory");
}
__device__ __forceinline__ void wait_asynccnt0() {
#if __has_builtin(__builtin_amdgcn_s_wait_asynccnt)
  __builtin_amdgcn_s_wait_asynccnt(0);
#else
  asm volatile("s_wait_asynccnt 0x0" ::: "memory");
#endif
}
// flat shared-memory address -> LDS byte offset (low 32 bits of LDS aperture)
__device__ __forceinline__ unsigned lds_off(const void* p) {
  return (unsigned)(unsigned long long)p;
}

// ---------------------------------------------------------------- converters
__global__ __launch_bounds__(256) void f32_to_f16_kernel(const float* __restrict__ x,
                                                         _Float16* __restrict__ y, int n) {
  int i = blockIdx.x * 256 + threadIdx.x;
  if (i < n) y[i] = (_Float16)x[i];
}

// ---------------------------------------------------------------- layernorm (C = 256)
// one wave per token; lane l owns channels l, l+32, ..., l+224
__global__ __launch_bounds__(256) void layernorm_kernel(const float* __restrict__ x,
                                                        const float* __restrict__ g,
                                                        const float* __restrict__ b,
                                                        _Float16* __restrict__ y) {
  const int wave = threadIdx.x >> 5, lane = threadIdx.x & 31;
  const size_t t = (size_t)blockIdx.x * 8 + wave;
  const float* xr = x + t * 256;
  float v[8], s = 0.f, ss = 0.f;
#pragma unroll
  for (int i = 0; i < 8; ++i) {
    v[i] = xr[lane + 32 * i];
    s += v[i];
    ss += v[i] * v[i];
  }
#pragma unroll
  for (int m = 16; m >= 1; m >>= 1) {
    s  += __shfl_xor(s,  m, 32);
    ss += __shfl_xor(ss, m, 32);
  }
  const float mu  = s * (1.f / 256.f);
  const float var = ss * (1.f / 256.f) - mu * mu;
  const float rs  = rsqrtf(var + 1e-5f);
  _Float16* yr = y + t * 256;
#pragma unroll
  for (int i = 0; i < 8; ++i) {
    int c = lane + 32 * i;
    yr[c] = (_Float16)((v[i] - mu) * rs * g[c] + b[c]);
  }
}

// ---------------------------------------------------------------- per-token 8x8 head softmax
// qkv row = 768 halfs = [q(8x32) | k(8x32) | v(8x32)]; one wave per token
__global__ __launch_bounds__(256) void attn_heads_kernel(const _Float16* __restrict__ qkv,
                                                         _Float16* __restrict__ out) {
  __shared__ _Float16 qs[8][768];
  __shared__ float    sc[8][64];
  const int wave = threadIdx.x >> 5, lane = threadIdx.x & 31;
  const size_t t = (size_t)blockIdx.x * 8 + wave;

  const uint4* src = (const uint4*)(qkv + t * 768);   // 96 16B chunks
  uint4* dst = (uint4*)(&qs[wave][0]);
#pragma unroll
  for (int i = 0; i < 3; ++i) dst[lane + 32 * i] = src[lane + 32 * i];
  __syncthreads();

  for (int s = lane; s < 64; s += 32) {
    const int h = s >> 3, gg = s & 7;
    const _Float16* qp = &qs[wave][h * 32];
    const _Float16* kp = &qs[wave][256 + gg * 32];
    float a = 0.f;
#pragma unroll
    for (int d = 0; d < 32; ++d) a += (float)qp[d] * (float)kp[d];
    sc[wave][s] = a * 0.17677669529663687f;   // 1/sqrt(32)
  }
  __syncthreads();

  if (lane < 8) {
    float mx = sc[wave][lane * 8];
#pragma unroll
    for (int gg = 1; gg < 8; ++gg) mx = fmaxf(mx, sc[wave][lane * 8 + gg]);
    float e[8], sum = 0.f;
#pragma unroll
    for (int gg = 0; gg < 8; ++gg) { e[gg] = __expf(sc[wave][lane * 8 + gg] - mx); sum += e[gg]; }
    const float inv = 1.f / sum;
#pragma unroll
    for (int gg = 0; gg < 8; ++gg) sc[wave][lane * 8 + gg] = e[gg] * inv;
  }
  __syncthreads();

#pragma unroll
  for (int h = 0; h < 8; ++h) {
    float a = 0.f;
#pragma unroll
    for (int gg = 0; gg < 8; ++gg) a += sc[wave][h * 8 + gg] * (float)qs[wave][512 + gg * 32 + lane];
    out[t * 256 + h * 32 + lane] = (_Float16)a;
  }
}

// ---------------------------------------------------------------- WMMA GEMM
// D[M,N] = A[M,K] * B[N,K]^T + bias  (optional ReLU, optional +resid, f16 or f32 out)
// 256 threads = 8 waves arranged 4x2; WG tile 128x128; wave tile 32x64
// (2 A frags, 4 B frags, 8 accumulators, 8 WMMAs per K-step of 32).
// Double-buffered LDS, one barrier per K-step; next tile staged with
// global_load_async_to_lds_b128 (ASYNCcnt DMA) so it hides under the WMMAs.
template <int RELU, int RESID, int OUTF16>
__global__ __launch_bounds__(256) void gemm_wmma_kernel(const _Float16* __restrict__ A,
                                                        const _Float16* __restrict__ B,
                                                        const float* __restrict__ bias,
                                                        const float* __restrict__ resid,
                                                        _Float16* __restrict__ outH,
                                                        float* __restrict__ outF,
                                                        int M, int N, int K) {
  __shared__ _Float16 As[2][128 * LDSTR];
  __shared__ _Float16 Bs[2][128 * LDSTR];
  const int tid = threadIdx.x, lane = tid & 31, wave = tid >> 5;
  const int rowBase = blockIdx.y * 128, colBase = blockIdx.x * 128;
  const int rowGrp = wave >> 1;   // 0..3 -> rows rowGrp*32
  const int colGrp = wave & 1;    // 0..1 -> cols colGrp*64

  v8f acc[2][4];
#pragma unroll
  for (int i = 0; i < 2; ++i)
#pragma unroll
    for (int j = 0; j < 4; ++j)
      acc[i][j] = (v8f){0.f, 0.f, 0.f, 0.f, 0.f, 0.f, 0.f, 0.f};

  const int mrow = lane & 15;   // A row / B col within 16
  const int kh   = lane >> 4;   // lane half selects K sub-range

  // per-thread staging coords: tile is 128 rows x 32 halfs = 512 x 16B chunks,
  // thread handles chunks tid and tid+256 -> rows (tid>>2) and (tid>>2)+64
  const int gr = tid >> 2, gko = (tid & 3) * 8;
  const _Float16* Ag0 = A + (size_t)(rowBase + gr)      * K + gko;
  const _Float16* Ag1 = A + (size_t)(rowBase + gr + 64) * K + gko;
  const _Float16* Bg0 = B + (size_t)(colBase + gr)      * K + gko;
  const _Float16* Bg1 = B + (size_t)(colBase + gr + 64) * K + gko;
  const int lA0 = gr * LDSTR + gko, lA1 = (gr + 64) * LDSTR + gko;

  const int nIter = K / 32;

  // prologue: DMA tile 0 into buffer 0
  async_load_b128(lds_off(&As[0][lA0]), Ag0);
  async_load_b128(lds_off(&As[0][lA1]), Ag1);
  async_load_b128(lds_off(&Bs[0][lA0]), Bg0);
  async_load_b128(lds_off(&Bs[0][lA1]), Bg1);

  for (int it = 0; it < nIter; ++it) {
    wait_asynccnt0();       // this wave's DMA into buf[cur] has landed
    __syncthreads();        // all waves' DMA landed; buf[nxt] readers are done
    const int cur = it & 1, nxt = cur ^ 1;
    const bool more = (it + 1) < nIter;
    if (more) {
      const int k0 = (it + 1) * 32;
      async_load_b128(lds_off(&As[nxt][lA0]), Ag0 + k0);
      async_load_b128(lds_off(&As[nxt][lA1]), Ag1 + k0);
      async_load_b128(lds_off(&Bs[nxt][lA0]), Bg0 + k0);
      async_load_b128(lds_off(&Bs[nxt][lA1]), Bg1 + k0);
    }

    // A fragments (16-bit A 16x32 layout, ISA 7.12.2): lanes0-15 K{0..7,16..23}, lanes16-31 +8
    v16h af[2];
#pragma unroll
    for (int ma = 0; ma < 2; ++ma) {
      const _Float16* base = &As[cur][(rowGrp * 32 + ma * 16 + mrow) * LDSTR];
#pragma unroll
      for (int j = 0; j < 8; ++j) {
        int kk = 2 * j + 8 * kh + ((j >= 4) ? 8 : 0);
        af[ma][2 * j]     = base[kk];
        af[ma][2 * j + 1] = base[kk + 1];
      }
    }
    // B fragments: same K order as A: lanes0-15 K{0..7,16..23}, lanes16-31 +8
    v16h bf[4];
#pragma unroll
    for (int nt = 0; nt < 4; ++nt) {
      const _Float16* base = &Bs[cur][(colGrp * 64 + nt * 16 + mrow) * LDSTR];
#pragma unroll
      for (int j = 0; j < 8; ++j) {
        int kk = 2 * j + 8 * kh + ((j >= 4) ? 8 : 0);
        bf[nt][2 * j]     = base[kk];
        bf[nt][2 * j + 1] = base[kk + 1];
      }
    }
    // 8 back-to-back WMMAs (independent accumulators -> no hazard NOPs)
#pragma unroll
    for (int ma = 0; ma < 2; ++ma)
#pragma unroll
      for (int nt = 0; nt < 4; ++nt)
        acc[ma][nt] = __builtin_amdgcn_wmma_f32_16x16x32_f16(
            false, af[ma], false, bf[nt], (short)0, acc[ma][nt], false, false);
  }

  // epilogue: C/D layout — VGPR r, lanes0-15 -> M=r, lanes16-31 -> M=r+8; N = lane&15
#pragma unroll
  for (int ma = 0; ma < 2; ++ma) {
#pragma unroll
    for (int nt = 0; nt < 4; ++nt) {
      const int col = colBase + colGrp * 64 + nt * 16 + mrow;
      const float bv = bias[col];
#pragma unroll
      for (int r = 0; r < 8; ++r) {
        const int row = rowBase + rowGrp * 32 + ma * 16 + r + 8 * kh;
        float v = acc[ma][nt][r] + bv;
        if (RELU) v = fmaxf(v, 0.f);
        if (RESID) v += resid[(size_t)row * N + col];
        if (OUTF16) outH[(size_t)row * N + col] = (_Float16)v;
        else        outF[(size_t)row * N + col] = v;
      }
    }
  }
}

// ---------------------------------------------------------------- launcher
extern "C" void kernel_launch(void* const* d_in, const int* in_sizes, int n_in,
                              void* d_out, int out_size, void* d_ws, size_t ws_size,
                              hipStream_t stream) {
  const float* x      = (const float*)d_in[0];
  const float* ln1_g  = (const float*)d_in[1];
  const float* ln1_b  = (const float*)d_in[2];
  const float* qkv_w  = (const float*)d_in[3];   // [768,256]
  const float* qkv_b  = (const float*)d_in[4];
  const float* proj_w = (const float*)d_in[5];   // [256,256]
  const float* proj_b = (const float*)d_in[6];
  const float* ln2_g  = (const float*)d_in[7];
  const float* ln2_b  = (const float*)d_in[8];
  const float* ffn_w1 = (const float*)d_in[9];   // [1024,256]
  const float* ffn_b1 = (const float*)d_in[10];
  const float* ffn_w2 = (const float*)d_in[11];  // [256,1024]
  const float* ffn_b2 = (const float*)d_in[12];

  const int M = 16 * 64 * 64;   // 65536 tokens

  // workspace layout (bytes); H aliases the dead QKV+ATTN region
  char* ws = (char*)d_ws;
  _Float16* Y     = (_Float16*)(ws);                                   // 33,554,432 (Y1 / Y2)
  _Float16* QKV   = (_Float16*)(ws + 33554432ULL);                     // 100,663,296
  _Float16* ATTN  = (_Float16*)(ws + 33554432ULL + 100663296ULL);      // 33,554,432
  _Float16* Hbuf  = (_Float16*)(ws + 33554432ULL);                     // 134,217,728 (alias)
  float*    X1    = (float*)(ws + 167772160ULL);                       // 67,108,864
  char*     wp    = ws + 234881024ULL;
  _Float16* WqkvH  = (_Float16*)(wp);
  _Float16* WprojH = (_Float16*)(wp + 393216ULL);
  _Float16* W1H    = (_Float16*)(wp + 393216ULL + 131072ULL);
  _Float16* W2H    = (_Float16*)(wp + 393216ULL + 131072ULL + 524288ULL);

  // weights -> f16 (re-done every call: deterministic, poison-safe)
  f32_to_f16_kernel<<<(196608 + 255) / 256, 256, 0, stream>>>(qkv_w,  WqkvH,  196608);
  f32_to_f16_kernel<<<(65536  + 255) / 256, 256, 0, stream>>>(proj_w, WprojH, 65536);
  f32_to_f16_kernel<<<(262144 + 255) / 256, 256, 0, stream>>>(ffn_w1, W1H,    262144);
  f32_to_f16_kernel<<<(262144 + 255) / 256, 256, 0, stream>>>(ffn_w2, W2H,    262144);

  // LN1
  layernorm_kernel<<<M / 8, 256, 0, stream>>>(x, ln1_g, ln1_b, Y);
  // QKV: [M,256] @ [768,256]^T -> f16
  gemm_wmma_kernel<0, 0, 1><<<dim3(768 / 128, M / 128), 256, 0, stream>>>(
      Y, WqkvH, qkv_b, nullptr, QKV, nullptr, M, 768, 256);
  // per-token 8x8 head softmax
  attn_heads_kernel<<<M / 8, 256, 0, stream>>>(QKV, ATTN);
  // proj + residual(x) -> X1 (f32)
  gemm_wmma_kernel<0, 1, 0><<<dim3(256 / 128, M / 128), 256, 0, stream>>>(
      ATTN, WprojH, proj_b, x, nullptr, X1, M, 256, 256);
  // LN2
  layernorm_kernel<<<M / 8, 256, 0, stream>>>(X1, ln2_g, ln2_b, Y);
  // FFN1 + ReLU -> f16 H
  gemm_wmma_kernel<1, 0, 1><<<dim3(1024 / 128, M / 128), 256, 0, stream>>>(
      Y, W1H, ffn_b1, nullptr, Hbuf, nullptr, M, 1024, 256);
  // FFN2 + residual(X1) -> d_out (f32)
  gemm_wmma_kernel<0, 1, 0><<<dim3(256 / 128, M / 128), 256, 0, stream>>>(
      Hbuf, W2H, ffn_b2, X1, nullptr, (float*)d_out, M, 256, 1024);
}
